// SycophancyDetectorLite_31250182045927
// MI455X (gfx1250) — compile-verified, matrix operand order checked by reading, not run on hardware
//
#include <hip/hip_runtime.h>
#include <hip/hip_bf16.h>

#define NU 50000
#define NA 50000
#define DD 384
#define HH 4
#define DHD 96
#define EE 200000
#define MPAD 50048   // NU/NA padded to multiple of 128 for fragment-tiled A

typedef __attribute__((ext_vector_type(16))) __bf16 v16bf;
typedef __attribute__((ext_vector_type(2)))  __bf16 v2bf;
typedef __attribute__((ext_vector_type(8)))  float  v8f;

__device__ __forceinline__ float gelu_exact(float x) {
    return 0.5f * x * (1.0f + erff(x * 0.70710678118f));
}

// ---- monotonic float<->uint encoding for atomicMax over floats ----
__device__ __forceinline__ unsigned f2mono(float f) {
    unsigned u = __float_as_uint(f);
    return (u & 0x80000000u) ? ~u : (u | 0x80000000u);
}
__device__ __forceinline__ float mono2f(unsigned u) {
    return (u & 0x80000000u) ? __uint_as_float(u & 0x7fffffffu)
                             : __uint_as_float(~u);
}

// =================== utility kernels ===================
__global__ void fill_f32_kernel(float* p, float v, long n) {
    long i = (long)blockIdx.x * blockDim.x + threadIdx.x;
    if (i < n) p[i] = v;
}
__global__ void fill_u32_kernel(unsigned* p, unsigned v, long n) {
    long i = (long)blockIdx.x * blockDim.x + threadIdx.x;
    if (i < n) p[i] = v;
}

// fp32 [K,N] row-major -> bf16 WMMA B-fragment order:
//   frag (nt,kt): 32 lanes x 16 bf16 (32B/lane); lane=(n%16)+16*half(k),
//   half=(k>>3)&1, j=(k&7)+8*(k>>4); frag index = nt*(K/32)+kt
__global__ void cvt_swizzle_B_kernel(const float* __restrict__ src, __bf16* __restrict__ dst,
                                     int K, int N, int nmat)
{
    long total = (long)nmat * K * N;
    long i = (long)blockIdx.x * blockDim.x + threadIdx.x;
    if (i >= total) return;
    int kn  = K * N;
    int mat = (int)(i / kn);
    int rem = (int)(i % kn);
    int k = rem / N, n = rem % N;
    int half = (k >> 3) & 1;
    int j    = (k & 7) + 8 * (k >> 4);
    int lane = (n & 15) + 16 * half;
    int nt = n >> 4, kt = k >> 5;
    long di = (long)mat * kn + (((long)(nt * (K >> 5) + kt)) * 32 + lane) * 16 + j;
    dst[di] = (__bf16)src[i];
}

// fp32 [M,K] row-major (+optional exact-gelu) -> bf16 WMMA A-fragment order, M padded:
//   frag (mslab,kt): lane=(m%16)+16*half(k), j as above; frag index = (m>>4)*(K/32)+kt
__global__ void cvt_swizzle_A_kernel(const float* __restrict__ src, __bf16* __restrict__ dst,
                                     int M, int K, int Mpad, int act)
{
    long total = (long)Mpad * K;
    long i = (long)blockIdx.x * blockDim.x + threadIdx.x;
    if (i >= total) return;
    int m = (int)(i / K), k = (int)(i % K);
    float v = 0.f;
    if (m < M) {
        v = src[(long)m * K + k];
        if (act) v = gelu_exact(v);
    }
    int half = (k >> 3) & 1;
    int j    = (k & 7) + 8 * (k >> 4);
    int lane = (m & 15) + 16 * half;
    long di = (((long)(m >> 4) * (K >> 5) + (k >> 5)) * 32 + lane) * 16 + j;
    dst[di] = (__bf16)v;
}

// =================== WMMA GEMM, both operands pre-swizzled: C = A @ B + bias ===================
// 256 threads (8 waves), tile 128(M) x 64(N). No LDS, no barriers:
// per k-step each wave does 1 A-fragment load + 4 B-fragment loads + 4 WMMAs.
__global__ void gemm_bias_kernel(const __bf16* __restrict__ Af,  // fragment-ordered, Mpad rows
                                 const __bf16* __restrict__ Bf,  // fragment-ordered
                                 const float* __restrict__ bias,
                                 float* __restrict__ C,
                                 int M, int N, int K)
{
    const int tid  = threadIdx.x;
    const int lane = tid & 31;
    const int wave = tid >> 5;            // 0..7
    const int hlf  = lane >> 4;
    const int lidx = lane & 15;
    const int m0 = blockIdx.x * 128;
    const int n0 = blockIdx.y * 64;
    const v16bf* Av = (const v16bf*)Af;
    const v16bf* Bv = (const v16bf*)Bf;
    const int ktiles = K >> 5;
    const long aslab = (long)((m0 >> 4) + wave) * ktiles;

    v8f acc[4];
    #pragma unroll
    for (int i = 0; i < 4; ++i) acc[i] = (v8f){0,0,0,0,0,0,0,0};

    for (int kt = 0; kt < ktiles; ++kt) {
        v16bf af = Av[(aslab + kt) * 32 + lane];
        #pragma unroll
        for (int nt = 0; nt < 4; ++nt) {
            v16bf bfrag = Bv[((long)((n0 >> 4) + nt) * ktiles + kt) * 32 + lane];
            acc[nt] = __builtin_amdgcn_wmma_f32_16x16x32_bf16(
                false, af, false, bfrag, (short)0, acc[nt], false, false);
        }
    }

    #pragma unroll
    for (int nt = 0; nt < 4; ++nt) {
        #pragma unroll
        for (int r = 0; r < 8; ++r) {
            int gm = m0 + wave * 16 + r + 8 * hlf;
            int gn = n0 + nt * 16 + lidx;
            if (gm < M) C[(long)gm * N + gn] = acc[nt][r] + bias[gn];
        }
    }
}

// =================== edge pass 1: alpha = (Q_dst[dst] . (K_src[src]@att_h)) * pri/sqrt(DH) ===================
// 128 threads (4 waves), 64 edges per block, blockIdx.y = head. Gathered A staged in LDS fragment order.
__global__ void edge_alpha_kernel(const float* __restrict__ Ksrc,
                                  const float* __restrict__ Qdst,
                                  const int* __restrict__ src,
                                  const int* __restrict__ dst,
                                  const __bf16* __restrict__ attf, // fragment-ordered [H][96*96]
                                  const float* __restrict__ pri,
                                  float* __restrict__ alpha,
                                  int nedges)
{
    __shared__ v16bf sAf[4][32];
    __shared__ float sK[64][96];
    const int tid  = threadIdx.x;
    const int lane = tid & 31, wave = tid >> 5;
    const int hlf  = lane >> 4, lidx = lane & 15;
    const int h  = blockIdx.y;
    const int e0 = blockIdx.x * 64;
    const v16bf* Bv = (const v16bf*)(attf + (long)h * DHD * DHD);

    v8f acc[6];
    #pragma unroll
    for (int i = 0; i < 6; ++i) acc[i] = (v8f){0,0,0,0,0,0,0,0};

    for (int kk = 0; kk < DHD; kk += 32) {
        #pragma unroll
        for (int it = 0; it < 8; ++it) {
            int p  = tid + 128 * it;      // 0..1023
            int r  = p >> 4;              // edge row 0..63
            int c0 = (p & 15) << 1;
            int e = e0 + r; if (e >= nedges) e = nedges - 1;
            int sr = src[e];
            const float* ap = Ksrc + (long)sr * DD + h * DHD + kk + c0;
            float v0 = ap[0], v1 = ap[1];
            int half = (c0 >> 3) & 1;
            int j    = (c0 & 7) + 8 * (c0 >> 4);
            int ln   = (r & 15) + 16 * half;
            v2bf pk; pk.x = (__bf16)v0; pk.y = (__bf16)v1;
            *(__shared__ v2bf*)((__shared__ __bf16*)&sAf[r >> 4][ln] + j) = pk;
        }
        __syncthreads();
        v16bf af = sAf[wave][lane];
        const int kt = kk >> 5;
        #pragma unroll
        for (int nt = 0; nt < 6; ++nt) {
            v16bf bfrag = Bv[(nt * 3 + kt) * 32 + lane];
            acc[nt] = __builtin_amdgcn_wmma_f32_16x16x32_bf16(
                false, af, false, bfrag, (short)0, acc[nt], false, false);
        }
        __syncthreads();
    }

    #pragma unroll
    for (int nt = 0; nt < 6; ++nt)
        #pragma unroll
        for (int r = 0; r < 8; ++r)
            sK[wave * 16 + r + 8 * hlf][nt * 16 + lidx] = acc[nt][r];
    __syncthreads();

    if (tid < 64) {
        int e = e0 + tid;
        if (e < nedges) {
            int d = dst[e];
            const float4* q  = (const float4*)(Qdst + (long)d * DD + h * DHD);
            const float4* kv = (const float4*)(&sK[tid][0]);
            float s = 0.f;
            #pragma unroll
            for (int f = 0; f < DHD / 4; ++f) {
                float4 a = kv[f], b = q[f];
                s += a.x * b.x + a.y * b.y + a.z * b.z + a.w * b.w;
            }
            alpha[(long)e * HH + h] = s * pri[h] * 0.10206207262f; // 1/sqrt(96)
        }
    }
}

// =================== segment softmax helpers ===================
__global__ void seg_max_kernel(const float* __restrict__ alpha, const int* __restrict__ dst,
                               unsigned* __restrict__ m, int nedges)
{
    long i = (long)blockIdx.x * blockDim.x + threadIdx.x;
    if (i < (long)nedges * HH) {
        int e = (int)(i >> 2), h = (int)(i & 3);
        atomicMax(&m[(long)dst[e] * HH + h], f2mono(alpha[i]));
    }
}
__global__ void seg_expsum_kernel(float* __restrict__ alpha, const int* __restrict__ dst,
                                  const unsigned* __restrict__ m, float* __restrict__ ssum,
                                  int nedges)
{
    long i = (long)blockIdx.x * blockDim.x + threadIdx.x;
    if (i < (long)nedges * HH) {
        int e = (int)(i >> 2), h = (int)(i & 3);
        float mm = mono2f(m[(long)dst[e] * HH + h]);
        float ee = expf(alpha[i] - mm);
        alpha[i] = ee;
        atomicAdd(&ssum[(long)dst[e] * HH + h], ee);
    }
}

// =================== edge pass 2: aggr[dst] += (V_src[src]@msg_h) * w[e,h] ===================
__global__ void edge_scatter_kernel(const float* __restrict__ Vsrc,
                                    const int* __restrict__ src,
                                    const int* __restrict__ dst,
                                    const __bf16* __restrict__ msgf,  // fragment-ordered
                                    const float* __restrict__ ealpha,
                                    const float* __restrict__ ssum,
                                    float* __restrict__ aggr,
                                    int nedges)
{
    __shared__ v16bf sAf[4][32];
    __shared__ float sW[64];
    __shared__ int   sD[64];
    const int tid  = threadIdx.x;
    const int lane = tid & 31, wave = tid >> 5;
    const int hlf  = lane >> 4, lidx = lane & 15;
    const int h  = blockIdx.y;
    const int e0 = blockIdx.x * 64;
    const v16bf* Bv = (const v16bf*)(msgf + (long)h * DHD * DHD);

    v8f acc[6];
    #pragma unroll
    for (int i = 0; i < 6; ++i) acc[i] = (v8f){0,0,0,0,0,0,0,0};

    if (tid < 64) {
        int e = e0 + tid;
        if (e < nedges) {
            int d = dst[e];
            sD[tid] = d;
            sW[tid] = ealpha[(long)e * HH + h] / (ssum[(long)d * HH + h] + 1e-16f);
        } else { sD[tid] = 0; sW[tid] = 0.f; }
    }

    for (int kk = 0; kk < DHD; kk += 32) {
        #pragma unroll
        for (int it = 0; it < 8; ++it) {
            int p  = tid + 128 * it;
            int r  = p >> 4;
            int c0 = (p & 15) << 1;
            int e = e0 + r; if (e >= nedges) e = nedges - 1;
            int sr = src[e];
            const float* ap = Vsrc + (long)sr * DD + h * DHD + kk + c0;
            float v0 = ap[0], v1 = ap[1];
            int half = (c0 >> 3) & 1;
            int j    = (c0 & 7) + 8 * (c0 >> 4);
            int ln   = (r & 15) + 16 * half;
            v2bf pk; pk.x = (__bf16)v0; pk.y = (__bf16)v1;
            *(__shared__ v2bf*)((__shared__ __bf16*)&sAf[r >> 4][ln] + j) = pk;
        }
        __syncthreads();
        v16bf af = sAf[wave][lane];
        const int kt = kk >> 5;
        #pragma unroll
        for (int nt = 0; nt < 6; ++nt) {
            v16bf bfrag = Bv[(nt * 3 + kt) * 32 + lane];
            acc[nt] = __builtin_amdgcn_wmma_f32_16x16x32_bf16(
                false, af, false, bfrag, (short)0, acc[nt], false, false);
        }
        __syncthreads();
    }

    #pragma unroll
    for (int nt = 0; nt < 6; ++nt) {
        #pragma unroll
        for (int r = 0; r < 8; ++r) {
            int widx = wave * 16 + r + 8 * hlf;
            int e = e0 + widx;
            if (e < nedges) {
                int f = nt * 16 + lidx;
                atomicAdd(&aggr[(long)sD[widx] * DD + h * DHD + f], acc[nt][r] * sW[widx]);
            }
        }
    }
}

// =================== blend + gelu + LN + mean-pool ===================
__global__ void blend_ln_pool_kernel(const float* __restrict__ htmp,
                                     const float* __restrict__ x,
                                     const float* __restrict__ skip,
                                     const float* __restrict__ ln_g,
                                     const float* __restrict__ ln_b,
                                     float* __restrict__ pool,
                                     int nrows)
{
    __shared__ float red[128];
    const int tid = threadIdx.x;   // 128 threads, 3 cols each
    const float sk = 1.f / (1.f + expf(-skip[0]));
    float pacc[3] = {0.f, 0.f, 0.f};

    for (int row = blockIdx.x; row < nrows; row += gridDim.x) {
        float g[3];
        float lsum = 0.f;
        #pragma unroll
        for (int j = 0; j < 3; ++j) {
            int c = tid + 128 * j;
            float hv = sk * htmp[(long)row * DD + c] + (1.f - sk) * x[(long)row * DD + c];
            float gv = gelu_exact(hv);
            g[j] = gv; lsum += gv;
        }
        red[tid] = lsum; __syncthreads();
        for (int s = 64; s > 0; s >>= 1) { if (tid < s) red[tid] += red[tid + s]; __syncthreads(); }
        float mu = red[0] * (1.f / 384.f); __syncthreads();
        float vs = 0.f;
        #pragma unroll
        for (int j = 0; j < 3; ++j) { float d = g[j] - mu; vs += d * d; }
        red[tid] = vs; __syncthreads();
        for (int s = 64; s > 0; s >>= 1) { if (tid < s) red[tid] += red[tid + s]; __syncthreads(); }
        float var = red[0] * (1.f / 384.f); __syncthreads();
        float rstd = rsqrtf(var + 1e-5f);
        #pragma unroll
        for (int j = 0; j < 3; ++j) {
            int c = tid + 128 * j;
            float y = (g[j] - mu) * rstd * ln_g[c] + ln_b[c];
            pacc[j] += y;
        }
    }
    const float invn = 1.f / (float)nrows;
    #pragma unroll
    for (int j = 0; j < 3; ++j) atomicAdd(&pool[tid + 128 * j], pacc[j] * invn);
}

// =================== final MLP head ===================
__global__ void final_mlp_kernel(const float* __restrict__ pool,  // [768] means
                                 const float* __restrict__ w1,    // [768,384]
                                 const float* __restrict__ b1,
                                 const float* __restrict__ w2,    // [384]
                                 const float* __restrict__ b2,
                                 float* __restrict__ out)
{
    __shared__ float red[384];
    const int j = threadIdx.x;     // 384 threads
    float acc = b1[j];
    for (int i = 0; i < 2 * DD; ++i) acc += pool[i] * w1[i * DD + j];
    float h1 = acc > 0.f ? acc : 0.f;
    red[j] = h1 * w2[j];
    __syncthreads();
    if (j == 0) {
        float t = 0.f;
        for (int i = 0; i < DD; ++i) t += red[i];
        out[0] = t + b2[0];
    }
}

// =================== host launcher ===================
extern "C" void kernel_launch(void* const* d_in, const int* in_sizes, int n_in,
                              void* d_out, int out_size, void* d_ws, size_t ws_size,
                              hipStream_t stream)
{
    (void)in_sizes; (void)n_in; (void)out_size; (void)ws_size;
    const float* x_user = (const float*)d_in[0];
    const float* x_ai   = (const float*)d_in[1];
    const int* e_u2a = (const int*)d_in[2];
    const int* e_a2u = (const int*)d_in[3];
    const int *src_u2a = e_u2a,     *dst_u2a = e_u2a + EE;
    const int *src_a2u = e_a2u,     *dst_a2u = e_a2u + EE;
    const float* kb_u = (const float*)d_in[5];
    const float* qb_u = (const float*)d_in[7];
    const float* vb_u = (const float*)d_in[9];
    const float* ab_u = (const float*)d_in[11];
    const float* skip_u = (const float*)d_in[12];
    const float* kb_a = (const float*)d_in[14];
    const float* qb_a = (const float*)d_in[16];
    const float* vb_a = (const float*)d_in[18];
    const float* ab_a = (const float*)d_in[20];
    const float* skip_a = (const float*)d_in[21];
    const float* pri_ua = (const float*)d_in[24];
    const float* pri_au = (const float*)d_in[27];
    const float* ln_g = (const float*)d_in[28];
    const float* ln_b = (const float*)d_in[29];
    const float* w1 = (const float*)d_in[30];
    const float* b1 = (const float*)d_in[31];
    const float* w2 = (const float*)d_in[32];
    const float* b2 = (const float*)d_in[33];

    // ---- workspace carve-up ----
    char* ws = (char*)d_ws;
    size_t off = 0;
    auto carve = [&](size_t bytes) -> char* {
        char* p = ws + off;
        off += (bytes + 255) & ~(size_t)255;
        return p;
    };
    __bf16* wb[8];
    for (int i = 0; i < 8; ++i) wb[i] = (__bf16*)carve((size_t)DD * DD * sizeof(__bf16));
    __bf16* attb[4];
    for (int i = 0; i < 4; ++i) attb[i] = (__bf16*)carve((size_t)HH * DHD * DHD * sizeof(__bf16));
    __bf16* xs_u = (__bf16*)carve((size_t)MPAD * DD * sizeof(__bf16));  // swizzled A (user)
    __bf16* xs_a = (__bf16*)carve((size_t)MPAD * DD * sizeof(__bf16));  // swizzled A (ai)
    float* Ku = (float*)carve((size_t)NU * DD * 4);
    float* Qu = (float*)carve((size_t)NU * DD * 4);
    float* Vu = (float*)carve((size_t)NU * DD * 4);
    float* Ka = (float*)carve((size_t)NA * DD * 4);
    float* Qa = (float*)carve((size_t)NA * DD * 4);
    float* Va = (float*)carve((size_t)NA * DD * 4);
    float* alpha_ua = (float*)carve((size_t)EE * HH * 4);
    float* alpha_au = (float*)carve((size_t)EE * HH * 4);
    unsigned* m_a = (unsigned*)carve((size_t)NA * HH * 4);
    float*    s_a = (float*)carve((size_t)NA * HH * 4);
    unsigned* m_u = (unsigned*)carve((size_t)NU * HH * 4);
    float*    s_u = (float*)carve((size_t)NU * HH * 4);
    float* aggr_a = (float*)carve((size_t)NA * DD * 4);
    float* aggr_u = (float*)carve((size_t)NU * DD * 4);
    float* pool   = (float*)carve(2 * DD * 4);
    float* htmp_u = Ku;   // K/Q/V dead after edge passes -> reuse
    float* htmp_a = Ka;

    const long nW = (long)DD * DD;
    const long nAtt = (long)HH * DHD * DHD;
    const long nA = (long)MPAD * DD;

    // ---- 1) convert + swizzle weights / relation matrices to bf16 fragment order ----
    const int widx[8] = {4, 6, 8, 10, 13, 15, 17, 19};
    for (int i = 0; i < 8; ++i)
        cvt_swizzle_B_kernel<<<(nW + 255) / 256, 256, 0, stream>>>(
            (const float*)d_in[widx[i]], wb[i], DD, DD, 1);
    const int aidx[4] = {22, 23, 25, 26};
    for (int i = 0; i < 4; ++i)
        cvt_swizzle_B_kernel<<<(nAtt + 255) / 256, 256, 0, stream>>>(
            (const float*)d_in[aidx[i]], attb[i], DHD, DHD, HH);

    // ---- 2) swizzle activations (A operands) ----
    cvt_swizzle_A_kernel<<<(nA + 255) / 256, 256, 0, stream>>>(x_user, xs_u, NU, DD, MPAD, 0);
    cvt_swizzle_A_kernel<<<(nA + 255) / 256, 256, 0, stream>>>(x_ai,   xs_a, NA, DD, MPAD, 0);

    // ---- 3) init accumulators ----
    {
        long n;
        n = (long)NA * HH; fill_u32_kernel<<<(n+255)/256,256,0,stream>>>(m_a, 0u, n);
        n = (long)NU * HH; fill_u32_kernel<<<(n+255)/256,256,0,stream>>>(m_u, 0u, n);
        n = (long)NA * HH; fill_f32_kernel<<<(n+255)/256,256,0,stream>>>(s_a, 0.f, n);
        n = (long)NU * HH; fill_f32_kernel<<<(n+255)/256,256,0,stream>>>(s_u, 0.f, n);
        n = (long)NA * DD; fill_f32_kernel<<<(n+255)/256,256,0,stream>>>(aggr_a, 0.f, n);
        n = (long)NU * DD; fill_f32_kernel<<<(n+255)/256,256,0,stream>>>(aggr_u, 0.f, n);
        n = 2 * DD;        fill_f32_kernel<<<(n+255)/256,256,0,stream>>>(pool, 0.f, n);
    }

    // ---- 4) K/Q/V projections (pure-fragment WMMA GEMMs) ----
    dim3 gG(MPAD / 128, DD / 64);
    gemm_bias_kernel<<<gG, 256, 0, stream>>>(xs_u, wb[0], kb_u, Ku, NU, DD, DD);
    gemm_bias_kernel<<<gG, 256, 0, stream>>>(xs_u, wb[1], qb_u, Qu, NU, DD, DD);
    gemm_bias_kernel<<<gG, 256, 0, stream>>>(xs_u, wb[2], vb_u, Vu, NU, DD, DD);
    gemm_bias_kernel<<<gG, 256, 0, stream>>>(xs_a, wb[4], kb_a, Ka, NA, DD, DD);
    gemm_bias_kernel<<<gG, 256, 0, stream>>>(xs_a, wb[5], qb_a, Qa, NA, DD, DD);
    gemm_bias_kernel<<<gG, 256, 0, stream>>>(xs_a, wb[6], vb_a, Va, NA, DD, DD);

    // ---- 5) edge attention (user -> ai), then (ai -> user) ----
    dim3 gE((EE + 63) / 64, HH);
    long nEH = (long)EE * HH;
    int gSeg = (int)((nEH + 255) / 256);

    edge_alpha_kernel<<<gE, 128, 0, stream>>>(Ku, Qa, src_u2a, dst_u2a, attb[0], pri_ua, alpha_ua, EE);
    seg_max_kernel<<<gSeg, 256, 0, stream>>>(alpha_ua, dst_u2a, m_a, EE);
    seg_expsum_kernel<<<gSeg, 256, 0, stream>>>(alpha_ua, dst_u2a, m_a, s_a, EE);
    edge_scatter_kernel<<<gE, 128, 0, stream>>>(Vu, src_u2a, dst_u2a, attb[1], alpha_ua, s_a, aggr_a, EE);

    edge_alpha_kernel<<<gE, 128, 0, stream>>>(Ka, Qu, src_a2u, dst_a2u, attb[2], pri_au, alpha_au, EE);
    seg_max_kernel<<<gSeg, 256, 0, stream>>>(alpha_au, dst_a2u, m_u, EE);
    seg_expsum_kernel<<<gSeg, 256, 0, stream>>>(alpha_au, dst_a2u, m_u, s_u, EE);
    edge_scatter_kernel<<<gE, 128, 0, stream>>>(Va, src_a2u, dst_a2u, attb[3], alpha_au, s_u, aggr_u, EE);

    // ---- 6) post projection: gelu(aggr) @ aw + ab (gelu fused into the A swizzle pass) ----
    cvt_swizzle_A_kernel<<<(nA + 255) / 256, 256, 0, stream>>>(aggr_u, xs_u, NU, DD, MPAD, 1);
    cvt_swizzle_A_kernel<<<(nA + 255) / 256, 256, 0, stream>>>(aggr_a, xs_a, NA, DD, MPAD, 1);
    gemm_bias_kernel<<<gG, 256, 0, stream>>>(xs_u, wb[3], ab_u, htmp_u, NU, DD, DD);
    gemm_bias_kernel<<<gG, 256, 0, stream>>>(xs_a, wb[7], ab_a, htmp_a, NA, DD, DD);

    // ---- 7) skip-blend + gelu + LN + mean pool ----
    blend_ln_pool_kernel<<<512, 128, 0, stream>>>(htmp_u, x_user, skip_u, ln_g, ln_b, pool,      NU);
    blend_ln_pool_kernel<<<512, 128, 0, stream>>>(htmp_a, x_ai,   skip_a, ln_g, ln_b, pool + DD, NA);

    // ---- 8) MLP head ----
    final_mlp_kernel<<<1, DD, 0, stream>>>(pool, w1, b1, w2, b2, (float*)d_out);
}
